// PowerPredictor_85925115724500
// MI455X (gfx1250) — matrix operand order repaired and checked by reading, not compiled, not run on hardware
//
#include <hip/hip_runtime.h>
#include <cstdint>
#include <cstddef>

// ---------------------------------------------------------------------------
// CDNA5 / gfx1250: wave32, WMMA bf16 16x16x32 with f32 accumulation,
// async global->LDS weight staging (ASYNCcnt path).
// ---------------------------------------------------------------------------
typedef __attribute__((ext_vector_type(16))) __bf16 v16bf;
typedef __attribute__((ext_vector_type(8)))  float  v8f;

static inline int div_up(long a, long b) { return (int)((a + b - 1) / b); }

__device__ inline float sigf(float x) { return 1.0f / (1.0f + __expf(-x)); }

// ---------------- zero ----------------
__global__ void k_zero(float* p, size_t n) {
  size_t i = (size_t)blockIdx.x * blockDim.x + threadIdx.x;
  if (i < n) p[i] = 0.f;
}

// ---------------- edge message + scatter-add (GNN aggregation) ----------------
// msg = relu(h[src] + edge_attr*We + be); aggr[dst] += msg
__global__ void k_edge_scatter(const float* __restrict__ h, const float* __restrict__ ea,
                               const int* __restrict__ src, const int* __restrict__ dst,
                               const float* __restrict__ We, const float* __restrict__ be,
                               float* __restrict__ aggr, int E, int d) {
  int idx = blockIdx.x * blockDim.x + threadIdx.x;
  int e = idx / d;
  if (e >= E) return;
  int j = idx - e * d;
  float m = h[(size_t)src[e] * d + j] + ea[e] * We[j] + be[j];
  if (m > 0.f) atomicAdd(&aggr[(size_t)dst[e] * d + j], m);
}

// ---------------- z = h + aggr -> bf16 (K-padded for WMMA) ----------------
__global__ void k_addconv(const float* __restrict__ h, const float* __restrict__ aggr,
                          __bf16* __restrict__ zb, int M, int d, int Kpad) {
  int idx = blockIdx.x * blockDim.x + threadIdx.x;
  int i = idx / Kpad;
  if (i >= M) return;
  int k = idx - i * Kpad;
  float v = (k < d) ? (h[(size_t)i * d + k] + aggr[(size_t)i * d + k]) : 0.f;
  zb[idx] = (__bf16)v;
}

// ---------------- pack weights: f32 [Kin][N] -> bf16 transposed [N][Kpad] ----------------
__global__ void k_packWT(const float* __restrict__ W, __bf16* __restrict__ Wt,
                         int Kin, int Kpad, int N) {
  int idx = blockIdx.x * blockDim.x + threadIdx.x;
  if (idx >= N * Kpad) return;
  int n = idx / Kpad, k = idx - n * Kpad;
  Wt[idx] = (__bf16)((k < Kin) ? W[(size_t)k * N + n] : 0.f);
}

// ---------------- WMMA GEMM: out[M x 128] = A[M x KT] @ W + bias ----------------
// A bf16 row-major (streamed from HBM), Wt bf16 pre-transposed [128][KT] staged
// once per workgroup into LDS via async global->LDS copies (ASYNCcnt), then read
// as B fragments with ds_load_b128. One wave per 16-row M tile, 8 accumulators
// cover all 128 output columns. EPI 0: relu -> bf16; EPI 1: raw f32.
template <int KT, int EPI>
__global__ __launch_bounds__(256) void k_gemm(const __bf16* __restrict__ A,
                                              const __bf16* __restrict__ Wt,
                                              const float* __restrict__ bias,
                                              void* __restrict__ outp, int Mtiles) {
  __shared__ __bf16 wlds[128 * KT];
  // --- stage weights: 128*KT*2 bytes, 16B per lane per async op ---
  {
    const int bytes = 128 * KT * 2;
    for (int off = threadIdx.x * 16; off < bytes; off += 256 * 16) {
      unsigned int ldsa = (unsigned int)(size_t)((char*)wlds + off);
      unsigned long long ga = (unsigned long long)(size_t)((const char*)Wt + off);
      asm volatile("global_load_async_to_lds_b128 %0, %1, off"
                   :: "v"(ldsa), "v"(ga) : "memory");
    }
    asm volatile("s_wait_asynccnt 0x0" ::: "memory");
  }
  __syncthreads();

  int wv = threadIdx.x >> 5;
  int mtile = blockIdx.x * 8 + wv;
  if (mtile < Mtiles) {
    int lane = threadIdx.x & 31, l15 = lane & 15, hf = lane >> 4;
    v8f acc[8] = {};
    const __bf16* arow = A + (size_t)(mtile * 16 + l15) * KT;
    for (int kk = 0; kk < KT; kk += 32) {
      // A fragment 16x32: lane half selects K sub-blocks {0..15} / {16..31}
      union { v16bf v; uint4 u[2]; } af;
      af.u[0] = *(const uint4*)(arow + kk + hf * 16);
      af.u[1] = *(const uint4*)(arow + kk + hf * 16 + 8);
#pragma unroll
      for (int nt = 0; nt < 8; ++nt) {
        // B fragment 32x16 from LDS: lane holds column (l15), 16 consecutive K values
        union { v16bf v; uint4 u[2]; } bfm;
        const __bf16* bp = wlds + (size_t)(nt * 16 + l15) * KT + kk + hf * 16;
        bfm.u[0] = *(const uint4*)(bp);
        bfm.u[1] = *(const uint4*)(bp + 8);
        acc[nt] = __builtin_amdgcn_wmma_f32_16x16x32_bf16(
            false, af.v, false, bfm.v, (short)0, acc[nt], false, false);
      }
    }
#pragma unroll
    for (int nt = 0; nt < 8; ++nt) {
      int col = nt * 16 + l15;
      float bv = bias[col];
#pragma unroll
      for (int r = 0; r < 8; ++r) {
        int row = mtile * 16 + r + 8 * hf;
        float v = acc[nt][r] + bv;
        if constexpr (EPI == 0) {
          v = fmaxf(v, 0.f);
          ((__bf16*)outp)[(size_t)row * 128 + col] = (__bf16)v;
        } else {
          ((float*)outp)[(size_t)row * 128 + col] = v;
        }
      }
    }
  }
}

// ---------------- LayerNorm(128) + ReLU, one wave per row ----------------
__global__ void k_ln_relu(const float* __restrict__ y, const float* __restrict__ g,
                          const float* __restrict__ b, float* __restrict__ h, int M) {
  int row = blockIdx.x * 8 + (threadIdx.x >> 5);
  if (row >= M) return;
  int lane = threadIdx.x & 31;
  const float* yr = y + (size_t)row * 128;
  float v[4], s = 0.f, s2 = 0.f;
#pragma unroll
  for (int i = 0; i < 4; i++) { v[i] = yr[lane + 32 * i]; s += v[i]; s2 += v[i] * v[i]; }
#pragma unroll
  for (int o = 16; o > 0; o >>= 1) { s += __shfl_xor(s, o, 32); s2 += __shfl_xor(s2, o, 32); }
  float mu = s / 128.f;
  float var = s2 / 128.f - mu * mu;
  float rs = rsqrtf(var + 1e-5f);
  float* hr = h + (size_t)row * 128;
#pragma unroll
  for (int i = 0; i < 4; i++) {
    int c = lane + 32 * i;
    hr[c] = fmaxf((v[i] - mu) * rs * g[c] + b[c], 0.f);
  }
}

// ---------------- graph pooling (sum + count via atomics; batch ids contiguous) ----------------
__global__ void k_pool(const float* __restrict__ h, const int* __restrict__ batch,
                       float* __restrict__ gf, float* __restrict__ cnt, int Nn) {
  int idx = blockIdx.x * blockDim.x + threadIdx.x;
  if (idx >= Nn * 128) return;
  int i = idx >> 7, c = idx & 127;
  int bi = batch[i];
  atomicAdd(&gf[bi * 256 + 128 + c], h[idx]);  // sum pool in cols 128..255
  if (c == 0) atomicAdd(&cnt[bi], 1.f);
}
__global__ void k_pool_fin(float* gf, const float* cnt) {
  int idx = blockIdx.x * blockDim.x + threadIdx.x;
  if (idx >= 64 * 128) return;
  int b = idx >> 7, c = idx & 127;
  gf[b * 256 + c] = gf[b * 256 + 128 + c] / fmaxf(cnt[b], 1.f);  // mean pool cols 0..127
}

// ---------------- embedding lookup -> bf16 [T][64][64] ----------------
__global__ void k_embed(const int* __restrict__ seq, const float* __restrict__ emb,
                        __bf16* __restrict__ X) {
  int idx = blockIdx.x * blockDim.x + threadIdx.x;
  if (idx >= 50 * 64 * 64) return;
  int t = idx / (64 * 64);
  int rem = idx - t * 64 * 64;
  int b = rem >> 6, e = rem & 63;
  int tok = seq[b * 50 + t];
  X[idx] = (__bf16)emb[tok * 64 + e];
}

__global__ void k_f2b(const float* __restrict__ in, __bf16* __restrict__ out, int n) {
  int i = blockIdx.x * blockDim.x + threadIdx.x;
  if (i < n) out[i] = (__bf16)in[i];
}

// ---------------- pack LSTM weights: bf16 [512][K], gate-interleaved columns ----------------
// packed col n: gate g = n&3, unit u = n>>2 (orig col = g*128+u); K = din+128 ([Wih;Whh])
__global__ void k_pack_lstm(const float* __restrict__ Wih, const float* __restrict__ Whh,
                            const float* __restrict__ bih, const float* __restrict__ bhh,
                            __bf16* __restrict__ Wt, float* __restrict__ bsum, int din) {
  int K = din + 128;
  int idx = blockIdx.x * blockDim.x + threadIdx.x;
  if (idx >= 512 * K) return;
  int n = idx / K, k = idx - n * K;
  int g = n & 3, u = n >> 2;
  int co = g * 128 + u;
  float w = (k < din) ? Wih[(size_t)k * 512 + co] : Whh[(size_t)(k - din) * 512 + co];
  Wt[idx] = (__bf16)w;
  if (k == 0) bsum[n] = bih[co] + bhh[co];
}

// ---------------- bidirectional LSTM layer: one block per direction, 16 waves ----------------
// Each wave: M tile mt (16 batch rows), 128 gate columns (8 WMMA tiles). Gate columns are
// interleaved i/f/g/o per unit so a lane-quad holds all 4 gates of one (b,u) cell.
// h kept in LDS (bf16), c register-resident. out[t][b][dir*128+u] = masked h.
__global__ __launch_bounds__(512) void k_lstm(const __bf16* __restrict__ X,
                                              const __bf16* __restrict__ Wt,
                                              const float* __restrict__ bias,
                                              const int* __restrict__ lengths,
                                              float* __restrict__ out, int T, int din) {
  const int K = din + 128;
  int dir = blockIdx.x;
  const __bf16* W = Wt + (size_t)dir * 512 * K;
  const float* bs = bias + dir * 512;
  __shared__ __bf16 xbuf[64 * 256];
  __shared__ __bf16 hbuf[64 * 128];
  int tid = threadIdx.x;
  int wv = tid >> 5, lane = tid & 31, l15 = lane & 15, hf = lane >> 4;
  int mt = wv & 3, ng = wv >> 2;
  for (int i = tid; i < 64 * 128; i += 512) hbuf[i] = (__bf16)0.f;
  float c[8][8];
#pragma unroll
  for (int a = 0; a < 8; a++)
#pragma unroll
    for (int r = 0; r < 8; r++) c[a][r] = 0.f;
  int lenr[8];
  float bv[8];
#pragma unroll
  for (int r = 0; r < 8; r++) lenr[r] = lengths[mt * 16 + r + 8 * hf];
#pragma unroll
  for (int nt = 0; nt < 8; nt++) bv[nt] = bs[ng * 128 + nt * 16 + l15];
  __syncthreads();
  for (int step = 0; step < T; ++step) {
    int t = dir ? (T - 1 - step) : step;
    for (int i = tid; i < 64 * din; i += 512) xbuf[i] = X[(size_t)t * 64 * din + i];
    __syncthreads();
    v8f acc[8] = {};
    int arow = mt * 16 + l15;
    for (int kk = 0; kk < K; kk += 32) {
      const __bf16* asrc;
      int kloc;
      if (kk < din) { asrc = xbuf + arow * din; kloc = kk; }
      else          { asrc = hbuf + arow * 128; kloc = kk - din; }
      union { v16bf v; uint4 u[2]; } af;
      af.u[0] = *(const uint4*)(asrc + kloc + hf * 16);
      af.u[1] = *(const uint4*)(asrc + kloc + hf * 16 + 8);
#pragma unroll
      for (int nt = 0; nt < 8; ++nt) {
        union { v16bf v; uint4 u[2]; } bfm;
        const __bf16* bp = W + (size_t)(ng * 128 + nt * 16 + l15) * K + kk + hf * 16;
        bfm.u[0] = *(const uint4*)bp;
        bfm.u[1] = *(const uint4*)(bp + 8);
        acc[nt] = __builtin_amdgcn_wmma_f32_16x16x32_bf16(
            false, af.v, false, bfm.v, (short)0, acc[nt], false, false);
      }
    }
    __syncthreads();  // all hbuf reads done before epilogue writes
    int q = lane & 3;
#pragma unroll
    for (int nt = 0; nt < 8; ++nt) {
#pragma unroll
      for (int r = 0; r < 8; ++r) {
        float v = acc[nt][r] + bv[nt];
        int baseL = lane & ~3;
        float gi = __shfl(v, baseL + 0, 32);
        float gfv = __shfl(v, baseL + 1, 32);
        float gg = __shfl(v, baseL + 2, 32);
        float go = __shfl(v, baseL + 3, 32);
        float cn = sigf(gfv) * c[nt][r] + sigf(gi) * tanhf(gg);
        float hn = sigf(go) * tanhf(cn);
        bool m = t < lenr[r];
        if (m) c[nt][r] = cn;
        if (q == 0) {
          int u = ng * 32 + nt * 4 + (l15 >> 2);
          int rowb = mt * 16 + r + 8 * hf;
          float heff = m ? hn : (float)hbuf[rowb * 128 + u];
          hbuf[rowb * 128 + u] = (__bf16)heff;
          out[(size_t)t * 64 * 256 + rowb * 256 + dir * 128 + u] = m ? hn : 0.f;
        }
      }
    }
    __syncthreads();
  }
}

// ---------------- attention pooling + gate + MLP head (one block per batch row) ----------------
__global__ void k_head(const float* __restrict__ out2, const int* __restrict__ seq,
                       const float* __restrict__ gf,
                       const float* __restrict__ attnW, const float* __restrict__ attnB,
                       const float* __restrict__ gateW, const float* __restrict__ gateB,
                       const float* __restrict__ h1W, const float* __restrict__ h1B,
                       const float* __restrict__ ln1g, const float* __restrict__ ln1b,
                       const float* __restrict__ h2W, const float* __restrict__ h2B,
                       const float* __restrict__ ln2g, const float* __restrict__ ln2b,
                       const float* __restrict__ h3W, const float* __restrict__ h3B,
                       float* __restrict__ outp) {
  int b = blockIdx.x;
  int tid = threadIdx.x;  // 256
  __shared__ float sw[64];
  __shared__ float rvec[256], zvec[512], z1[256], z2[128];
  __shared__ float red[2];
  if (tid < 50) {
    const float* o = out2 + ((size_t)tid * 64 + b) * 256;
    float s = attnB[0];
    for (int c = 0; c < 256; c++) s += o[c] * attnW[c];
    if (seq[b * 50 + tid] == 0) s = -1e9f;
    sw[tid] = s;
  }
  __syncthreads();
  if (tid == 0) {
    float mx = -1e30f;
    for (int t = 0; t < 50; t++) mx = fmaxf(mx, sw[t]);
    float sm = 0.f;
    for (int t = 0; t < 50; t++) { sw[t] = __expf(sw[t] - mx); sm += sw[t]; }
    for (int t = 0; t < 50; t++) sw[t] /= sm;
  }
  __syncthreads();
  {
    float a = 0.f;
    for (int t = 0; t < 50; t++) a += sw[t] * out2[((size_t)t * 64 + b) * 256 + tid];
    rvec[tid] = a;
    zvec[tid] = gf[b * 256 + tid];
  }
  __syncthreads();
  {
    float s = gateB[tid];
    for (int k = 0; k < 256; k++) s += zvec[k] * gateW[k * 256 + tid];
    for (int k = 0; k < 256; k++) s += rvec[k] * gateW[(256 + k) * 256 + tid];
    zvec[256 + tid] = rvec[tid] * sigf(s);
  }
  __syncthreads();
  {
    float s = h1B[tid];
    for (int k = 0; k < 512; k++) s += zvec[k] * h1W[k * 256 + tid];
    z1[tid] = fmaxf(s, 0.f);
  }
  __syncthreads();
  if (tid == 0) {
    float mu = 0.f;
    for (int k = 0; k < 256; k++) mu += z1[k];
    mu /= 256.f;
    float v = 0.f;
    for (int k = 0; k < 256; k++) { float d = z1[k] - mu; v += d * d; }
    red[0] = mu;
    red[1] = rsqrtf(v / 256.f + 1e-5f);
  }
  __syncthreads();
  z1[tid] = (z1[tid] - red[0]) * red[1] * ln1g[tid] + ln1b[tid];
  __syncthreads();
  if (tid < 128) {
    float s = h2B[tid];
    for (int k = 0; k < 256; k++) s += z1[k] * h2W[k * 128 + tid];
    z2[tid] = fmaxf(s, 0.f);
  }
  __syncthreads();
  if (tid == 0) {
    float mu = 0.f;
    for (int k = 0; k < 128; k++) mu += z2[k];
    mu /= 128.f;
    float v = 0.f;
    for (int k = 0; k < 128; k++) { float d = z2[k] - mu; v += d * d; }
    red[0] = mu;
    red[1] = rsqrtf(v / 128.f + 1e-5f);
  }
  __syncthreads();
  if (tid < 128) z2[tid] = (z2[tid] - red[0]) * red[1] * ln2g[tid] + ln2b[tid];
  __syncthreads();
  if (tid == 0) {
    float s = h3B[0];
    for (int k = 0; k < 128; k++) s += z2[k] * h3W[k];
    outp[b] = s;
  }
}

// ---------------------------------------------------------------------------
// Input index map: JAX pytree (sorted dict keys) flattening of setup_inputs().
// ---------------------------------------------------------------------------
enum {
  IN_BATCH = 0, IN_EDGE_ATTR = 1, IN_EDGE_INDEX = 2, IN_LENGTHS = 3,
  IN_ATTN_W = 4, IN_ATTN_B = 5,
  IN_C0L1W = 6, IN_C0L1B = 7, IN_C0L2W = 8, IN_C0L2B = 9,
  IN_C1L1W = 10, IN_C1L1B = 11, IN_C1L2W = 12, IN_C1L2B = 13,
  IN_C2L1W = 14, IN_C2L1B = 15, IN_C2L2W = 16, IN_C2L2B = 17,
  IN_E0W = 18, IN_E0B = 19, IN_E1W = 20, IN_E1B = 21, IN_E2W = 22, IN_E2B = 23,
  IN_EMB = 24, IN_GATE_W = 25, IN_GATE_B = 26,
  IN_H1W = 27, IN_H1B = 28, IN_H2W = 29, IN_H2B = 30, IN_H3W = 31, IN_H3B = 32,
  IN_LN1B = 33, IN_LN1G = 34, IN_LN2B = 35, IN_LN2G = 36,
  IN_L0B_WHH = 37, IN_L0B_WIH = 38, IN_L0B_BHH = 39, IN_L0B_BIH = 40,
  IN_L0F_WHH = 41, IN_L0F_WIH = 42, IN_L0F_BHH = 43, IN_L0F_BIH = 44,
  IN_L1B_WHH = 45, IN_L1B_WIH = 46, IN_L1B_BHH = 47, IN_L1B_BIH = 48,
  IN_L1F_WHH = 49, IN_L1F_WIH = 50, IN_L1F_BHH = 51, IN_L1F_BIH = 52,
  IN_N0B = 53, IN_N0G = 54, IN_N1B = 55, IN_N1G = 56, IN_N2B = 57, IN_N2G = 58,
  IN_SEQ = 59, IN_X = 60
};

extern "C" void kernel_launch(void* const* d_in, const int* in_sizes, int n_in,
                              void* d_out, int out_size, void* d_ws, size_t ws_size,
                              hipStream_t stream) {
  const int Nn = 200000, E = 400000, T = 50;
  const int Mt = Nn / 16;  // 12500 M tiles

  const int* batch = (const int*)d_in[IN_BATCH];
  const float* edge_attr = (const float*)d_in[IN_EDGE_ATTR];
  const int* ei = (const int*)d_in[IN_EDGE_INDEX];
  const int* src = ei, * dst = ei + E;
  const int* lengths = (const int*)d_in[IN_LENGTHS];
  const int* seq = (const int*)d_in[IN_SEQ];
  const float* x = (const float*)d_in[IN_X];
  const float* emb = (const float*)d_in[IN_EMB];

  const float* l1W[3] = { (const float*)d_in[IN_C0L1W], (const float*)d_in[IN_C1L1W], (const float*)d_in[IN_C2L1W] };
  const float* l1B[3] = { (const float*)d_in[IN_C0L1B], (const float*)d_in[IN_C1L1B], (const float*)d_in[IN_C2L1B] };
  const float* l2W[3] = { (const float*)d_in[IN_C0L2W], (const float*)d_in[IN_C1L2W], (const float*)d_in[IN_C2L2W] };
  const float* l2B[3] = { (const float*)d_in[IN_C0L2B], (const float*)d_in[IN_C1L2B], (const float*)d_in[IN_C2L2B] };
  const float* eW[3] = { (const float*)d_in[IN_E0W], (const float*)d_in[IN_E1W], (const float*)d_in[IN_E2W] };
  const float* eB[3] = { (const float*)d_in[IN_E0B], (const float*)d_in[IN_E1B], (const float*)d_in[IN_E2B] };
  const float* nG[3] = { (const float*)d_in[IN_N0G], (const float*)d_in[IN_N1G], (const float*)d_in[IN_N2G] };
  const float* nB[3] = { (const float*)d_in[IN_N0B], (const float*)d_in[IN_N1B], (const float*)d_in[IN_N2B] };

  // ---- workspace layout ----
  char* ws = (char*)d_ws;
  size_t off = 0;
  auto alloc = [&](size_t bytes) {
    size_t o = off;
    off = (off + bytes + 255) & ~(size_t)255;
    return (void*)(ws + o);
  };
  float* buf_h = (float*)alloc((size_t)Nn * 128 * 4);
  float* buf_y = (float*)alloc((size_t)Nn * 128 * 4);  // aggr, then GEMM2 output
  __bf16* buf_zb = (__bf16*)alloc((size_t)Nn * 128 * 2);
  __bf16* buf_tb = (__bf16*)alloc((size_t)Nn * 128 * 2);
  __bf16* wtA = (__bf16*)alloc(128 * 128 * 2);
  __bf16* wtB = (__bf16*)alloc(128 * 128 * 2);
  __bf16* lw0 = (__bf16*)alloc((size_t)2 * 512 * 192 * 2);
  __bf16* lw1 = (__bf16*)alloc((size_t)2 * 512 * 384 * 2);
  float* lb0 = (float*)alloc(2 * 512 * 4);
  float* lb1 = (float*)alloc(2 * 512 * 4);
  __bf16* X1 = (__bf16*)alloc((size_t)T * 64 * 64 * 2);
  float* out1 = (float*)alloc((size_t)T * 64 * 256 * 4);
  __bf16* X2 = (__bf16*)alloc((size_t)T * 64 * 256 * 2);
  float* out2 = (float*)alloc((size_t)T * 64 * 256 * 4);
  float* gf = (float*)alloc(64 * 256 * 4);
  float* cnt = (float*)alloc(64 * 4);

  // ---- GNN: 3 message-passing layers ----
  for (int L = 0; L < 3; ++L) {
    const int d = (L == 0) ? 5 : 128;
    const int Kpad = (L == 0) ? 32 : 128;
    const float* hin = (L == 0) ? x : buf_h;

    k_zero<<<div_up((long)Nn * d, 256), 256, 0, stream>>>(buf_y, (size_t)Nn * d);
    k_edge_scatter<<<div_up((long)E * d, 256), 256, 0, stream>>>(
        hin, edge_attr, src, dst, eW[L], eB[L], buf_y, E, d);
    k_addconv<<<div_up((long)Nn * Kpad, 256), 256, 0, stream>>>(hin, buf_y, buf_zb, Nn, d, Kpad);
    k_packWT<<<div_up(128 * Kpad, 256), 256, 0, stream>>>(l1W[L], wtA, d, Kpad, 128);
    k_packWT<<<div_up(128 * 128, 256), 256, 0, stream>>>(l2W[L], wtB, 128, 128, 128);
    if (L == 0)
      k_gemm<32, 0><<<div_up(Mt, 8), 256, 0, stream>>>(buf_zb, wtA, l1B[L], buf_tb, Mt);
    else
      k_gemm<128, 0><<<div_up(Mt, 8), 256, 0, stream>>>(buf_zb, wtA, l1B[L], buf_tb, Mt);
    k_gemm<128, 1><<<div_up(Mt, 8), 256, 0, stream>>>(buf_tb, wtB, l2B[L], buf_y, Mt);
    k_ln_relu<<<div_up(Nn, 8), 256, 0, stream>>>(buf_y, nG[L], nB[L], buf_h, Nn);
  }

  // ---- pooling ----
  k_zero<<<div_up(64 * 256, 256), 256, 0, stream>>>(gf, 64 * 256);
  k_zero<<<1, 64, 0, stream>>>(cnt, 64);
  k_pool<<<div_up((long)Nn * 128, 256), 256, 0, stream>>>(buf_h, batch, gf, cnt, Nn);
  k_pool_fin<<<div_up(64 * 128, 256), 256, 0, stream>>>(gf, cnt);

  // ---- LSTM ----
  k_embed<<<div_up(50 * 64 * 64, 256), 256, 0, stream>>>(seq, emb, X1);
  // layer 0 (din=64, K=192): dir0=forward, dir1=backward
  k_pack_lstm<<<div_up(512 * 192, 256), 256, 0, stream>>>(
      (const float*)d_in[IN_L0F_WIH], (const float*)d_in[IN_L0F_WHH],
      (const float*)d_in[IN_L0F_BIH], (const float*)d_in[IN_L0F_BHH], lw0, lb0, 64);
  k_pack_lstm<<<div_up(512 * 192, 256), 256, 0, stream>>>(
      (const float*)d_in[IN_L0B_WIH], (const float*)d_in[IN_L0B_WHH],
      (const float*)d_in[IN_L0B_BIH], (const float*)d_in[IN_L0B_BHH],
      lw0 + (size_t)512 * 192, lb0 + 512, 64);
  k_lstm<<<2, 512, 0, stream>>>(X1, lw0, lb0, lengths, out1, T, 64);
  k_f2b<<<div_up(50 * 64 * 256, 256), 256, 0, stream>>>(out1, X2, 50 * 64 * 256);
  // layer 1 (din=256, K=384)
  k_pack_lstm<<<div_up(512 * 384, 256), 256, 0, stream>>>(
      (const float*)d_in[IN_L1F_WIH], (const float*)d_in[IN_L1F_WHH],
      (const float*)d_in[IN_L1F_BIH], (const float*)d_in[IN_L1F_BHH], lw1, lb1, 256);
  k_pack_lstm<<<div_up(512 * 384, 256), 256, 0, stream>>>(
      (const float*)d_in[IN_L1B_WIH], (const float*)d_in[IN_L1B_WHH],
      (const float*)d_in[IN_L1B_BIH], (const float*)d_in[IN_L1B_BHH],
      lw1 + (size_t)512 * 384, lb1 + 512, 256);
  k_lstm<<<2, 512, 0, stream>>>(X2, lw1, lb1, lengths, out2, T, 256);

  // ---- attention + gate + head ----
  k_head<<<64, 256, 0, stream>>>(
      out2, seq, gf,
      (const float*)d_in[IN_ATTN_W], (const float*)d_in[IN_ATTN_B],
      (const float*)d_in[IN_GATE_W], (const float*)d_in[IN_GATE_B],
      (const float*)d_in[IN_H1W], (const float*)d_in[IN_H1B],
      (const float*)d_in[IN_LN1G], (const float*)d_in[IN_LN1B],
      (const float*)d_in[IN_H2W], (const float*)d_in[IN_H2B],
      (const float*)d_in[IN_LN2G], (const float*)d_in[IN_LN2B],
      (const float*)d_in[IN_H3W], (const float*)d_in[IN_H3B],
      (float*)d_out);
}
